// evalRetina_41102837023243
// MI455X (gfx1250) — matrix operand order repaired and checked by reading, not compiled, not run on hardware
//
#include <hip/hip_runtime.h>
#include <stdint.h>

// ---------------- problem constants ----------------
#define NBATCH   32
#define NANCH    63504          // 24*14*9 + 48*28*9 + 96*56*9
#define NCLS     11
#define TOPK     512
#define NMS_T    0.5f
#define OBJ_T    0.4f

// per-level anchor counts: L0(32): 24x14, L1(16): 48x28, L2(8): 96x56
#define L0_END   3024
#define L1_END   15120

__constant__ float c_anchors[9][2] = {
    {0.0239749f, 0.0904281f}, {0.2569985f, 0.3775594f},
    {0.0125396f, 0.0429755f}, {0.0255381f, 0.0382968f},
    {0.0090712f, 0.0187574f}, {0.0456727f, 0.0533909f},
    {0.127638f,  0.1695816f}, {0.0189022f, 0.0215612f},
    {0.0669595f, 0.1037171f}};

typedef __attribute__((ext_vector_type(16))) _Float16 v16h;
typedef __attribute__((ext_vector_type(8)))  float    v8f;

__device__ __forceinline__ uint32_t f32_to_ord(float f) {
    uint32_t u = __float_as_uint(f);
    return (u & 0x80000000u) ? ~u : (u | 0x80000000u);
}
__device__ __forceinline__ float ord_to_f32(uint32_t o) {
    uint32_t u = (o & 0x80000000u) ? (o & 0x7FFFFFFFu) : ~o;
    return __uint_as_float(u);
}
__device__ __forceinline__ uint64_t u64max(uint64_t a, uint64_t b) { return a > b ? a : b; }

__device__ __forceinline__ void anchor_of(int a, float& gx, float& gy,
                                          float& aw, float& ah,
                                          float& sw, float& sh) {
    int base, W, H;
    if (a < L0_END)      { base = 0;      W = 24; H = 14; }
    else if (a < L1_END) { base = L0_END; W = 48; H = 28; }
    else                 { base = L1_END; W = 96; H = 56; }
    int r    = a - base;
    int cell = r / 9;
    int an   = r - cell * 9;
    gx = (float)(cell % W);
    gy = (float)(cell / W);
    aw = c_anchors[an][0];
    ah = c_anchors[an][1];
    sw = (float)W;
    sh = (float)H;
}

// ================= kernel 1: fused sigmoid/argmax -> sortable key =================
__global__ void k_score(const float* __restrict__ cls, uint64_t* __restrict__ keys) {
    int g = blockIdx.x * blockDim.x + threadIdx.x;
    if (g >= NBATCH * NANCH) return;
    const float* c = cls + (size_t)g * NCLS;
    float m = c[0];
    int   am = 0;
#pragma unroll
    for (int j = 1; j < NCLS; ++j) {
        float v = c[j];
        if (v > m) { m = v; am = j; }
    }
    float score = 1.0f / (1.0f + expf(-m));             // sigmoid of max logit == max prob
    bool  valid = (am > 0) && (score > OBJ_T);
    float ms    = valid ? score : -1.0f;
    uint32_t a  = (uint32_t)(g % NANCH);
    keys[g] = ((uint64_t)f32_to_ord(ms) << 32) | (uint32_t)(~a);  // ties -> smaller idx
}

// ================= kernel 2: streaming bitonic top-512 per batch =================
__global__ void __launch_bounds__(TOPK) k_topk(const uint64_t* __restrict__ keys,
                                               uint64_t* __restrict__ topk) {
    __shared__ uint64_t best[TOPK];
    __shared__ uint64_t tile[TOPK];
    const int b = blockIdx.x, t = threadIdx.x;
    const uint64_t* K = keys + (size_t)b * NANCH;

    best[t] = 0ull;
    __syncthreads();

    for (int tb = 0; tb < NANCH; tb += TOPK) {
        int a = tb + t;
        tile[t] = (a < NANCH) ? K[a] : 0ull;
        __syncthreads();
        // full bitonic sort of the tile, descending
        for (int k = 2; k <= TOPK; k <<= 1) {
            for (int j = k >> 1; j > 0; j >>= 1) {
                int x = t ^ j;
                if (x > t) {
                    uint64_t u = tile[t], v = tile[x];
                    bool desc = ((t & k) == 0);
                    if (desc ? (u < v) : (u > v)) { tile[t] = v; tile[x] = u; }
                }
                __syncthreads();
            }
        }
        // keep top-512 of (best ∪ tile): elementwise max vs reversed tile -> bitonic
        best[t] = u64max(best[t], tile[TOPK - 1 - t]);
        __syncthreads();
        // bitonic merge back to sorted-descending
        for (int j = TOPK >> 1; j > 0; j >>= 1) {
            int x = t ^ j;
            if (x > t) {
                uint64_t u = best[t], v = best[x];
                if (u < v) { best[t] = v; best[x] = u; }
            }
            __syncthreads();
        }
    }
    topk[(size_t)b * TOPK + t] = best[t];
}

// ================= kernel 3: decode + WMMA same-class + IoU matrix + greedy NMS ===
__global__ void __launch_bounds__(TOPK) k_nms(const float* __restrict__ cls,
                                              const float* __restrict__ boxp,
                                              const uint64_t* __restrict__ topk,
                                              float* __restrict__ out) {
    __shared__ float4   sRaw[TOPK];          // async-gathered raw box preds (8 KB)
    __shared__ float4   sBox[TOPK];          // decoded xyxy boxes           (8 KB)
    __shared__ int      sCls[TOPK];          // argmax labels                (2 KB)
    __shared__ float    sScore[TOPK];        // raw max prob                 (2 KB)
    __shared__ uint32_t sameW[TOPK * 16];    // supp. bit-matrix rows        (32 KB)
    __shared__ uint32_t keepW[16];           // 512-bit keep mask

    const int b = blockIdx.x, t = threadIdx.x;
    const int lane = t & 31, wave = t >> 5;
    const int half = lane >> 4, l16 = lane & 15;

    // ---- candidate fetch ----
    uint64_t key = topk[(size_t)b * TOPK + t];
    int      idx = (int)(~(uint32_t)key);                // anchor index (always valid: A >= 512)
    float    ms  = ord_to_f32((uint32_t)(key >> 32));    // masked score (-1.0 if invalid)
    bool     kv  = ms > -1.0f;

    // async gather of this candidate's raw box pred (16 B) straight into LDS,
    // overlapped with the class recompute below. (CDNA5 ASYNCcnt path)
    {
        uint32_t ldsA = (uint32_t)(uintptr_t)(&sRaw[t]);
        uint64_t ga   = (uint64_t)(uintptr_t)(boxp + ((size_t)b * NANCH + idx) * 4);
        asm volatile("global_load_async_to_lds_b128 %0, %1, off"
                     :: "v"(ldsA), "v"(ga) : "memory");
    }

    // recompute label + raw score for the 512 winners (cheap VALU while DMA flies)
    {
        const float* c = cls + ((size_t)b * NANCH + idx) * NCLS;
        float m = c[0];
        int   am = 0;
#pragma unroll
        for (int j = 1; j < NCLS; ++j) {
            float v = c[j];
            if (v > m) { m = v; am = j; }
        }
        sCls[t]   = am;
        sScore[t] = 1.0f / (1.0f + expf(-m));
    }

    // keep mask via wave ballot: word w == threads [32w, 32w+31] == wave w
    {
        uint32_t kb = (uint32_t)__ballot(kv);
        if (lane == 0) keepW[wave] = kb;
    }

    asm volatile("s_wait_asynccnt 0" ::: "memory");
    __syncthreads();

    // ---- decode box (exp-decode relative to anchor grid) ----
    {
        float4 r = sRaw[t];
        float gx, gy, aw, ah, sw, sh;
        anchor_of(idx, gx, gy, aw, ah, sw, sh);
        float cx = (expf(r.x) - 1e-6f + gx) / sw;
        float cy = (expf(r.y) - 1e-6f + gy) / sh;
        float w  = expf(r.z) * aw;
        float h  = expf(r.w) * ah;
        sBox[t] = make_float4(cx - 0.5f * w, cy - 0.5f * h,
                              cx + 0.5f * w, cy + 0.5f * h);
    }
    __syncthreads();

    // ---- same-class bit-matrix via WMMA: OneHot(kc) * OneHot(kc)^T ----
    // 512 "pair" units: pair p = (row block ti = p>>4, word wp = p&15) covering
    // column blocks tj = 2*wp and 2*wp+1. Each pair does 2 v_wmma (shared A),
    // converts D to bits with __ballot, and writes 16 complete 32-bit words with
    // plain ds_store (no atomics, no zero-init: every word written exactly once).
    {
        for (int p = wave; p < 32 * 16; p += 16) {
            const int ti = p >> 4, wp = p & 15;
            const int ca = sCls[ti * 16 + l16];          // row class for this lane (A)
            v16h av;
#pragma unroll
            for (int e = 0; e < 16; ++e) {
                // A 16x32 f16 layout: pairs {0..3}->K 0..7 (+8*half), {4..7}->K 16..23 (+8*half)
                int pi = e >> 1, ip = e & 1;
                int Ka = (pi < 4 ? 2 * pi : 16 + 2 * (pi - 4)) + 8 * half + ip;
                av[e] = (Ka == ca) ? (_Float16)1.0f : (_Float16)0.0f;
            }
            const v8f cacc = {0.f, 0.f, 0.f, 0.f, 0.f, 0.f, 0.f, 0.f};
            uint32_t bE[8], bO[8];
#pragma unroll
            for (int s = 0; s < 2; ++s) {                // even / odd column block
                const int tj = 2 * wp + s;
                const int cb = sCls[tj * 16 + l16];      // col class for this lane (B)
                v16h bv;
#pragma unroll
                for (int e = 0; e < 16; ++e) {
                    // B 32x16 f16 layout: e<8 -> K=e+8*half, e>=8 -> K=16+(e-8)+8*half, N=lane%16
                    int Kb = (e < 8 ? e : e + 8) + 8 * half;
                    bv[e] = (Kb == cb) ? (_Float16)1.0f : (_Float16)0.0f;
                }
                v8f d = __builtin_amdgcn_wmma_f32_16x16x32_f16(
                    false, av, false, bv, (short)0, cacc, false, false);
                // ballot: bits 0..15 -> row r (cols j=tj*16+0..15), bits 16..31 -> row r+8
#pragma unroll
                for (int r = 0; r < 8; ++r) {
                    uint32_t bb = (uint32_t)__ballot(d[r] > 0.5f);
                    if (s == 0) bE[r] = bb; else bO[r] = bb;
                }
            }
            // splice half-words into full 32-bit suppression words; lane l owns row ti*16+l
            uint32_t myword = 0;
#pragma unroll
            for (int r = 0; r < 8; ++r) {
                uint32_t wE = (bE[r] & 0xFFFFu) | (bO[r] << 16);           // row r
                uint32_t wO = (bE[r] >> 16) | (bO[r] & 0xFFFF0000u);       // row r+8
                if (lane == r)     myword = wE;
                if (lane == r + 8) myword = wO;
            }
            if (lane < 16) sameW[(ti * 16 + lane) * 16 + wp] = myword;
        }
    }
    __syncthreads();

    // ---- IoU filter: rows become (same-class & IoU>T & j>i) ----
    {
        float4 bi = sBox[t];
        float  a1 = (bi.z - bi.x) * (bi.w - bi.y);
#pragma unroll 1
        for (int w = 0; w < 16; ++w) {
            uint32_t s    = sameW[t * 16 + w];
            uint32_t bits = 0u;
            while (s) {
                int j = (w << 5) + __builtin_ctz(s);
                s &= s - 1;
                if (j <= t) continue;
                float4 bj = sBox[j];
                float lx = fmaxf(bi.x, bj.x), ly = fmaxf(bi.y, bj.y);
                float rx = fminf(bi.z, bj.z), ry = fminf(bi.w, bj.w);
                float iw = fmaxf(rx - lx, 0.0f), ih = fmaxf(ry - ly, 0.0f);
                float inter = iw * ih;
                float a2 = (bj.z - bj.x) * (bj.w - bj.y);
                float iou = inter / (a1 + a2 - inter);
                if (iou > NMS_T) bits |= 1u << (j & 31);
            }
            sameW[t * 16 + w] = bits;
        }
    }
    __syncthreads();

    // ---- greedy sequential suppression on one wave32 (16 lanes hold 512 bits) ----
    if (t < 32) {
        uint32_t kw = (t < 16) ? keepW[t] : 0u;
        for (int i = 0; i < TOPK - 1; ++i) {
            uint32_t wrd = __shfl(kw, i >> 5, 32);       // uniform broadcast of word holding bit i
            if ((wrd >> (i & 31)) & 1u) {
                if (t < 16) kw &= ~sameW[i * 16 + t];
            }
        }
        if (t < 16) keepW[t] = kw;
    }
    __syncthreads();

    // ---- emit [prob, x1, y1, x2, y2, cls-1] * keep ----
    {
        float kf = ((keepW[t >> 5] >> (t & 31)) & 1u) ? 1.0f : 0.0f;
        float4 bx = sBox[t];
        float* o = out + ((size_t)b * TOPK + t) * 6;
        o[0] = sScore[t] * kf;
        o[1] = bx.x * kf;
        o[2] = bx.y * kf;
        o[3] = bx.z * kf;
        o[4] = bx.w * kf;
        o[5] = ((float)sCls[t] - 1.0f) * kf;
    }
}

// ================================ launcher ================================
extern "C" void kernel_launch(void* const* d_in, const int* in_sizes, int n_in,
                              void* d_out, int out_size, void* d_ws, size_t ws_size,
                              hipStream_t stream) {
    const float* cls = (const float*)d_in[0];   // [32, 63504, 11] f32
    const float* box = (const float*)d_in[1];   // [32, 63504, 4]  f32
    float* out = (float*)d_out;                 // [32, 512, 6]    f32

    uint64_t* keys = (uint64_t*)d_ws;                                   // 16.26 MB
    uint64_t* topk = (uint64_t*)((char*)d_ws + (size_t)NBATCH * NANCH * 8);

    const int total = NBATCH * NANCH;
    k_score<<<(total + 255) / 256, 256, 0, stream>>>(cls, keys);
    k_topk<<<NBATCH, TOPK, 0, stream>>>(keys, topk);
    k_nms<<<NBATCH, TOPK, 0, stream>>>(cls, box, topk, out);
}
